// GINDecoder_78898549227825
// MI455X (gfx1250) — matrix operand order repaired and checked by reading, compiled, not run on hardware
//
#include <hip/hip_runtime.h>
#include <hip/hip_bf16.h>
#include <math.h>

typedef _Float16 half_t;
typedef __attribute__((ext_vector_type(16))) half_t v16h;
typedef __attribute__((ext_vector_type(8)))  float  v8f;

#define H 128
#define TD 128
#define WAVES 4          // 4 waves / 128 threads per block for GEMM kernels

// ---------------------------------------------------------------------------
// WMMA helpers (CDNA5 v_wmma_f32_16x16x32_f16, wave32)
// A fragment: 16x32 f16, lane = M (0..15) in each half-wave; lanes 0-15 hold
// K=k0..k0+15, lanes 16-31 hold K=k0+16..k0+31 (matches B fragment).
// ---------------------------------------------------------------------------
__device__ __forceinline__ v16h load_a_frag(const half_t* xs, int ldk, int k0) {
  const int lane = threadIdx.x & 31;
  const int m  = lane & 15;
  const int kh = (lane >> 4) & 1;
  const half_t* row = xs + m * ldk + k0 + 16 * kh;
  v16h a;
#pragma unroll
  for (int v = 0; v < 8; ++v) {
    const int kk = 2 * v;
    a[2 * v]     = row[kk];
    a[2 * v + 1] = row[kk + 1];
  }
  return a;
}

// B fragment: 32x16 f16 from transposed weights Wt[N][K] (N-major, f16).
// lane = N column (0..15); lanes 0-15 hold K=k0..k0+15, lanes 16-31 K=k0+16..k0+31.
__device__ __forceinline__ v16h load_b_frag(const half_t* Wt, int ldk, int n0, int k0) {
  const int lane = threadIdx.x & 31;
  const int n  = n0 + (lane & 15);
  const int kh = (lane >> 4) & 1;
  const half_t* p = Wt + (size_t)n * ldk + k0 + 16 * kh;
  v16h b;
#pragma unroll
  for (int v = 0; v < 8; ++v) {
    b[2 * v]     = p[2 * v];
    b[2 * v + 1] = p[2 * v + 1];
  }
  return b;
}

__device__ __forceinline__ v8f wmma16(v16h a, v16h b, v8f c) {
  return __builtin_amdgcn_wmma_f32_16x16x32_f16(false, a, false, b, (short)0, c,
                                                false, false);
}

// GEMM of one 16xK LDS tile against Wt (f16 [128][K]) -> acc[8] (16x128 D tile)
__device__ __forceinline__ void gemm_tile(const half_t* xs, int K,
                                          const half_t* Wt, v8f acc[8]) {
  const v8f z = {0.f, 0.f, 0.f, 0.f, 0.f, 0.f, 0.f, 0.f};
#pragma unroll
  for (int nt = 0; nt < 8; ++nt) acc[nt] = z;
  for (int k0 = 0; k0 < K; k0 += 32) {
    v16h a = load_a_frag(xs, K, k0);
#pragma unroll
    for (int nt = 0; nt < 8; ++nt) {
      v16h b = load_b_frag(Wt, K, nt * 16, k0);
      acc[nt] = wmma16(a, b, acc[nt]);
    }
  }
}

// ---------------------------------------------------------------------------
// Weight prep: float [K][N] -> f16 transposed [N][Kpad] (zero padded)
// ---------------------------------------------------------------------------
__global__ void wprep_kernel(const float* __restrict__ W, half_t* __restrict__ Wt,
                             int K, int N, int Kpad) {
  int idx = blockIdx.x * blockDim.x + threadIdx.x;
  if (idx >= N * Kpad) return;
  int n = idx / Kpad, k = idx - n * Kpad;
  float v = (k < K) ? W[(size_t)k * N + n] : 0.f;
  Wt[(size_t)n * Kpad + k] = (half_t)v;
}

__global__ void fill_zero_kernel(float* __restrict__ p, int n) {
  int i = blockIdx.x * blockDim.x + threadIdx.x;
  if (i < n) p[i] = 0.f;
}

// offsets[g] = exclusive prefix sum of num_atoms (G=100, trivial)
__global__ void offsets_kernel(const int* __restrict__ num_atoms,
                               int* __restrict__ off, int G) {
  if (blockIdx.x == 0 && threadIdx.x == 0) {
    int s = 0;
    for (int g = 0; g < G; ++g) { off[g] = s; s += num_atoms[g]; }
    off[G] = s;
  }
}

__global__ void n2g_kernel(const int* __restrict__ off, int* __restrict__ n2g,
                           int N, int G) {
  int i = blockIdx.x * blockDim.x + threadIdx.x;
  if (i >= N) return;
  int lo = 0, hi = G;
  while (lo + 1 < hi) {
    int mid = (lo + hi) >> 1;
    if (off[mid] <= i) lo = mid; else hi = mid;
  }
  n2g[i] = lo;
}

// ---------------------------------------------------------------------------
// Time embedding + fc_time MLP (G=100, tiny; plain VALU)
// ---------------------------------------------------------------------------
__global__ __launch_bounds__(128) void time_emb_kernel(
    const float* __restrict__ t,
    const float* __restrict__ w1, const float* __restrict__ b1,  // [128,512]
    const float* __restrict__ w2, const float* __restrict__ b2,  // [512,128]
    float* __restrict__ temb, int G) {
  __shared__ float emb[128];
  __shared__ float hid[512];
  const int g = blockIdx.x;
  if (g >= G) return;
  const int tid = threadIdx.x;
  const float tv = t[g];
  {
    int k = tid & 63;
    float freq = expf(-logf(10000.f) * (float)k / 63.f);
    float arg = tv * freq;
    emb[tid] = (tid >= 64) ? cosf(arg) : sinf(arg);
  }
  __syncthreads();
#pragma unroll
  for (int p = 0; p < 4; ++p) {
    int o = tid + p * 128;
    float s = b1[o];
    for (int k = 0; k < 128; ++k) s = fmaf(emb[k], w1[k * 512 + o], s);
    hid[o] = fmaxf(s, 0.f);
  }
  __syncthreads();
  {
    float s = b2[tid];
    for (int k = 0; k < 512; ++k) s = fmaf(hid[k], w2[k * 128 + tid], s);
    temb[(size_t)g * 128 + tid] = s;
  }
}

// ---------------------------------------------------------------------------
// edge_attr = (relu(d*w1+b1)) @ w2 + b2  -> f16 [E,128]
// One wave per 16 edges; WMMA for the 128x128 second layer.
// ---------------------------------------------------------------------------
__global__ __launch_bounds__(128) void edge_attr_kernel(
    const float* __restrict__ dist,
    const float* __restrict__ w1, const float* __restrict__ b1,
    const half_t* __restrict__ w2t, const float* __restrict__ b2,
    half_t* __restrict__ out, int E) {
  __shared__ half_t xs_all[WAVES][16 * 128];
  __shared__ float dv_all[WAVES][16];
  const int wave = threadIdx.x >> 5, lane = threadIdx.x & 31;
  half_t* xs = xs_all[wave];
  float* dv = dv_all[wave];
  const int e0 = (blockIdx.x * WAVES + wave) * 16;
  if (e0 >= E) return;
  if (lane < 16) dv[lane] = dist[e0 + lane];
  for (int idx = lane; idx < 16 * 128; idx += 32) {
    int e = idx >> 7, j = idx & 127;
    float h = fmaf(dv[e], w1[j], b1[j]);
    xs[idx] = (half_t)fmaxf(h, 0.f);
  }
  v8f acc[8];
  gemm_tile(xs, 128, w2t, acc);
  const int mh = lane >> 4, nc = lane & 15;
#pragma unroll
  for (int nt = 0; nt < 8; ++nt) {
    int col = nt * 16 + nc;
    float bb = b2[col];
#pragma unroll
    for (int r = 0; r < 8; ++r) {
      int e = e0 + r + 8 * mh;
      out[(size_t)e * 128 + col] = (half_t)(acc[nt][r] + bb);
    }
  }
}

// ---------------------------------------------------------------------------
// atom_emb = MLP2(atom_types[N,100]) -> f32 [N,128]; K padded 100 -> 128
// ---------------------------------------------------------------------------
__global__ __launch_bounds__(128) void atom_emb_kernel(
    const float* __restrict__ X,
    const half_t* __restrict__ w1t, const float* __restrict__ b1,
    const half_t* __restrict__ w2t, const float* __restrict__ b2,
    float* __restrict__ out, int N) {
  __shared__ half_t xs_all[WAVES][16 * 128];
  const int wave = threadIdx.x >> 5, lane = threadIdx.x & 31;
  half_t* xs = xs_all[wave];
  const int n0 = (blockIdx.x * WAVES + wave) * 16;
  if (n0 >= N) return;
  for (int idx = lane; idx < 16 * 128; idx += 32) {
    int r = idx >> 7, j = idx & 127;
    float v = (j < 100) ? X[(size_t)(n0 + r) * 100 + j] : 0.f;
    xs[idx] = (half_t)v;
  }
  v8f acc[8];
  gemm_tile(xs, 128, w1t, acc);
  const int mh = lane >> 4, nc = lane & 15;
#pragma unroll
  for (int nt = 0; nt < 8; ++nt) {
    int col = nt * 16 + nc;
    float bb = b1[col];
#pragma unroll
    for (int r = 0; r < 8; ++r)
      xs[(r + 8 * mh) * 128 + col] = (half_t)fmaxf(acc[nt][r] + bb, 0.f);
  }
  gemm_tile(xs, 128, w2t, acc);
#pragma unroll
  for (int nt = 0; nt < 8; ++nt) {
    int col = nt * 16 + nc;
    float bb = b2[col];
#pragma unroll
    for (int r = 0; r < 8; ++r)
      out[(size_t)(n0 + r + 8 * mh) * 128 + col] = acc[nt][r] + bb;
  }
}

// ---------------------------------------------------------------------------
// node = MLP2(concat[z, temb[g], atom_emb]) ; K = 384
// ---------------------------------------------------------------------------
__global__ __launch_bounds__(128) void node_feat_kernel(
    const float* __restrict__ z, const float* __restrict__ temb,
    const float* __restrict__ aemb, const int* __restrict__ n2g,
    const half_t* __restrict__ w1t, const float* __restrict__ b1,
    const half_t* __restrict__ w2t, const float* __restrict__ b2,
    float* __restrict__ out, int N) {
  __shared__ half_t xs_all[WAVES][16 * 384];  // 48 KB
  const int wave = threadIdx.x >> 5, lane = threadIdx.x & 31;
  half_t* xs = xs_all[wave];
  const int n0 = (blockIdx.x * WAVES + wave) * 16;
  if (n0 >= N) return;
  for (int r = 0; r < 16; ++r) {
    const int node = n0 + r;
    const int g = n2g[node];
    half_t* row = xs + r * 384;
    for (int j = lane; j < 384; j += 32) {
      float v;
      if (j < 128)       v = z[(size_t)node * 128 + j];
      else if (j < 256)  v = temb[(size_t)g * 128 + (j - 128)];
      else               v = aemb[(size_t)node * 128 + (j - 256)];
      row[j] = (half_t)v;
    }
  }
  v8f acc[8];
  gemm_tile(xs, 384, w1t, acc);
  const int mh = lane >> 4, nc = lane & 15;
#pragma unroll
  for (int nt = 0; nt < 8; ++nt) {
    int col = nt * 16 + nc;
    float bb = b1[col];
#pragma unroll
    for (int r = 0; r < 8; ++r)
      xs[(r + 8 * mh) * 128 + col] = (half_t)fmaxf(acc[nt][r] + bb, 0.f);
  }
  gemm_tile(xs, 128, w2t, acc);
#pragma unroll
  for (int nt = 0; nt < 8; ++nt) {
    int col = nt * 16 + nc;
    float bb = b2[col];
#pragma unroll
    for (int r = 0; r < 8; ++r)
      out[(size_t)(n0 + r + 8 * mh) * 128 + col] = acc[nt][r] + bb;
  }
}

// ---------------------------------------------------------------------------
// GINE message + scatter-add: agg[dst] += relu(node[src] + edge_attr)
// One 4-channel chunk per thread (float4 gather + 4 f32 atomics).
// ---------------------------------------------------------------------------
__global__ __launch_bounds__(256) void gine_message_kernel(
    const float* __restrict__ node, const half_t* __restrict__ eattr,
    const int* __restrict__ src, const int* __restrict__ dst,
    float* __restrict__ agg, int E) {
  int tid = blockIdx.x * blockDim.x + threadIdx.x;
  int e = tid >> 5;
  if (e >= E) return;
  int c = (tid & 31) * 4;
  int s = src[e], d = dst[e];
  const float4 nv = *(const float4*)(node + (size_t)s * 128 + c);
  const half_t* ea = eattr + (size_t)e * 128 + c;
  float m0 = fmaxf(nv.x + (float)ea[0], 0.f);
  float m1 = fmaxf(nv.y + (float)ea[1], 0.f);
  float m2 = fmaxf(nv.z + (float)ea[2], 0.f);
  float m3 = fmaxf(nv.w + (float)ea[3], 0.f);
  float* ap = agg + (size_t)d * 128 + c;
  atomicAdd(ap + 0, m0);
  atomicAdd(ap + 1, m1);
  atomicAdd(ap + 2, m2);
  atomicAdd(ap + 3, m3);
}

// ---------------------------------------------------------------------------
// GINE node update: out = MLP2(node+agg) (+relu if !last) + node
// ---------------------------------------------------------------------------
__global__ __launch_bounds__(128) void conv_update_kernel(
    const float* __restrict__ node, const float* __restrict__ agg,
    const half_t* __restrict__ w1t, const float* __restrict__ b1,
    const half_t* __restrict__ w2t, const float* __restrict__ b2,
    float* __restrict__ out, int N, int do_relu) {
  __shared__ half_t xs_all[WAVES][16 * 128];
  const int wave = threadIdx.x >> 5, lane = threadIdx.x & 31;
  half_t* xs = xs_all[wave];
  const int n0 = (blockIdx.x * WAVES + wave) * 16;
  if (n0 >= N) return;
  for (int idx = lane; idx < 16 * 128; idx += 32) {
    int r = idx >> 7, j = idx & 127;
    size_t o = (size_t)(n0 + r) * 128 + j;
    xs[idx] = (half_t)(node[o] + agg[o]);
  }
  v8f acc[8];
  gemm_tile(xs, 128, w1t, acc);
  const int mh = lane >> 4, nc = lane & 15;
#pragma unroll
  for (int nt = 0; nt < 8; ++nt) {
    int col = nt * 16 + nc;
    float bb = b1[col];
#pragma unroll
    for (int r = 0; r < 8; ++r)
      xs[(r + 8 * mh) * 128 + col] = (half_t)fmaxf(acc[nt][r] + bb, 0.f);
  }
  gemm_tile(xs, 128, w2t, acc);
#pragma unroll
  for (int nt = 0; nt < 8; ++nt) {
    int col = nt * 16 + nc;
    float bb = b2[col];
#pragma unroll
    for (int r = 0; r < 8; ++r) {
      size_t o = (size_t)(n0 + r + 8 * mh) * 128 + col;
      float h = acc[nt][r] + bb;
      if (do_relu) h = fmaxf(h, 0.f);
      out[o] = h + node[o];
    }
  }
}

// ---------------------------------------------------------------------------
// Host launch
// ---------------------------------------------------------------------------
extern "C" void kernel_launch(void* const* d_in, const int* in_sizes, int n_in,
                              void* d_out, int out_size, void* d_ws, size_t ws_size,
                              hipStream_t stream) {
  (void)n_in; (void)out_size; (void)ws_size;
  const float* z       = (const float*)d_in[0];
  const float* t       = (const float*)d_in[1];
  const float* atypes  = (const float*)d_in[2];
  const float* edist   = (const float*)d_in[3];
  const int*   eidx    = (const int*)  d_in[4];
  const int*   natoms  = (const int*)  d_in[5];
  const float* node_w1 = (const float*)d_in[6];
  const float* node_b1 = (const float*)d_in[7];
  const float* node_w2 = (const float*)d_in[8];
  const float* node_b2 = (const float*)d_in[9];
  const float* time_w1 = (const float*)d_in[10];
  const float* time_b1 = (const float*)d_in[11];
  const float* time_w2 = (const float*)d_in[12];
  const float* time_b2 = (const float*)d_in[13];
  const float* edge_w1 = (const float*)d_in[14];
  const float* edge_b1 = (const float*)d_in[15];
  const float* edge_w2 = (const float*)d_in[16];
  const float* edge_b2 = (const float*)d_in[17];
  const float* fcn_w1  = (const float*)d_in[18];
  const float* fcn_b1  = (const float*)d_in[19];
  const float* fcn_w2  = (const float*)d_in[20];
  const float* fcn_b2  = (const float*)d_in[21];
  const float* conv_w1 = (const float*)d_in[22];
  const float* conv_b1 = (const float*)d_in[23];
  const float* conv_w2 = (const float*)d_in[24];
  const float* conv_b2 = (const float*)d_in[25];

  const int N = in_sizes[0] / H;   // 20000
  const int G = in_sizes[1];       // 100
  const int E = in_sizes[3];       // 640000

  // workspace partition
  char* ws = (char*)d_ws;
  size_t off = 0;
  auto take = [&](size_t bytes) -> void* {
    void* p = ws + off;
    off = (off + bytes + 255) & ~(size_t)255;
    return p;
  };
  half_t* eattr   = (half_t*)take((size_t)E * H * sizeof(half_t));
  float*  node_a  = (float*) take((size_t)N * H * sizeof(float));
  float*  node_b  = (float*) take((size_t)N * H * sizeof(float));
  float*  agg     = (float*) take((size_t)N * H * sizeof(float));
  float*  aemb    = (float*) take((size_t)N * H * sizeof(float));
  float*  temb    = (float*) take((size_t)G * TD * sizeof(float));
  int*    offs    = (int*)   take((size_t)(G + 1) * sizeof(int));
  int*    n2g     = (int*)   take((size_t)N * sizeof(int));
  half_t* nw1t    = (half_t*)take((size_t)128 * 128 * sizeof(half_t));
  half_t* nw2t    = (half_t*)take((size_t)128 * 128 * sizeof(half_t));
  half_t* ew2t    = (half_t*)take((size_t)128 * 128 * sizeof(half_t));
  half_t* fw1t    = (half_t*)take((size_t)128 * 384 * sizeof(half_t));
  half_t* fw2t    = (half_t*)take((size_t)128 * 128 * sizeof(half_t));
  half_t* cw1t    = (half_t*)take((size_t)3 * 128 * 128 * sizeof(half_t));
  half_t* cw2t    = (half_t*)take((size_t)3 * 128 * 128 * sizeof(half_t));

  // 1) weight prep (f16, transposed, padded)
  const int TB = 256;
  wprep_kernel<<<(128 * 128 + TB - 1) / TB, TB, 0, stream>>>(node_w1, nw1t, 100, 128, 128);
  wprep_kernel<<<(128 * 128 + TB - 1) / TB, TB, 0, stream>>>(node_w2, nw2t, 128, 128, 128);
  wprep_kernel<<<(128 * 128 + TB - 1) / TB, TB, 0, stream>>>(edge_w2, ew2t, 128, 128, 128);
  wprep_kernel<<<(128 * 384 + TB - 1) / TB, TB, 0, stream>>>(fcn_w1, fw1t, 384, 128, 384);
  wprep_kernel<<<(128 * 128 + TB - 1) / TB, TB, 0, stream>>>(fcn_w2, fw2t, 128, 128, 128);
  for (int i = 0; i < 3; ++i) {
    wprep_kernel<<<(128 * 128 + TB - 1) / TB, TB, 0, stream>>>(
        conv_w1 + (size_t)i * 128 * 128, cw1t + (size_t)i * 128 * 128, 128, 128, 128);
    wprep_kernel<<<(128 * 128 + TB - 1) / TB, TB, 0, stream>>>(
        conv_w2 + (size_t)i * 128 * 128, cw2t + (size_t)i * 128 * 128, 128, 128, 128);
  }

  // 2) graph offsets + node->graph map
  offsets_kernel<<<1, 32, 0, stream>>>(natoms, offs, G);
  n2g_kernel<<<(N + TB - 1) / TB, TB, 0, stream>>>(offs, n2g, N, G);

  // 3) time embedding MLP (per graph)
  time_emb_kernel<<<G, 128, 0, stream>>>(t, time_w1, time_b1, time_w2, time_b2, temb, G);

  // 4) edge embedding (WMMA), stored f16
  {
    int tiles = (E + 15) / 16;
    int blocks = (tiles + WAVES - 1) / WAVES;
    edge_attr_kernel<<<blocks, 128, 0, stream>>>(edist, edge_w1, edge_b1, ew2t,
                                                 edge_b2, eattr, E);
  }

  // 5) atom embedding (WMMA)
  {
    int tiles = (N + 15) / 16;
    int blocks = (tiles + WAVES - 1) / WAVES;
    atom_emb_kernel<<<blocks, 128, 0, stream>>>(atypes, nw1t, node_b1, nw2t,
                                                node_b2, aemb, N);
  }

  // 6) fused node features (K=384, WMMA)
  {
    int tiles = (N + 15) / 16;
    int blocks = (tiles + WAVES - 1) / WAVES;
    node_feat_kernel<<<blocks, 128, 0, stream>>>(z, temb, aemb, n2g, fw1t, fcn_b1,
                                                 fw2t, fcn_b2, node_a, N);
  }

  // 7) 3 GINE conv layers
  const float* cur = node_a;
  float* outs[3] = { node_b, node_a, (float*)d_out };
  int msg_blocks = ((size_t)E * 32 + TB - 1) / TB;
  int upd_blocks = (((N + 15) / 16) + WAVES - 1) / WAVES;
  for (int i = 0; i < 3; ++i) {
    fill_zero_kernel<<<(N * H + TB - 1) / TB, TB, 0, stream>>>(agg, N * H);
    gine_message_kernel<<<msg_blocks, TB, 0, stream>>>(cur, eattr, eidx, eidx + E,
                                                       agg, E);
    conv_update_kernel<<<upd_blocks, 128, 0, stream>>>(
        cur, agg, cw1t + (size_t)i * 128 * 128, conv_b1 + (size_t)i * 128,
        cw2t + (size_t)i * 128 * 128, conv_b2 + (size_t)i * 128, outs[i], N,
        (i < 2) ? 1 : 0);
    cur = outs[i];
  }
}
